// GenerativeVelocityGNN_51573967290795
// MI455X (gfx1250) — matrix operand order repaired and checked by reading, not compiled, not run on hardware
//
#include <hip/hip_runtime.h>
#include <hip/hip_bf16.h>
#include <math.h>

// ---------------------------------------------------------------------------
// Types for CDNA5 WMMA (wave32): D(16x16 f32) = A(16x32 bf16) x B(32x16 bf16) + C
// ---------------------------------------------------------------------------
typedef __attribute__((ext_vector_type(16))) __bf16 v16bf;
typedef __attribute__((ext_vector_type(8)))  float  v8f;

union FragAB {
    v16bf v;
    uint4 q[2];
};

// ---------------------------------------------------------------------------
// Helpers
// ---------------------------------------------------------------------------
__device__ __forceinline__ unsigned f2key(float f) {
    unsigned u = __float_as_uint(f);
    return (u & 0x80000000u) ? ~u : (u | 0x80000000u);
}
__device__ __forceinline__ float key2f(unsigned k) {
    unsigned u = (k & 0x80000000u) ? (k & 0x7FFFFFFFu) : ~k;
    return __uint_as_float(u);
}
__device__ __forceinline__ float elu1(float x) {
    return x > 0.f ? x : expm1f(x);
}

// ---------------------------------------------------------------------------
// LDS-free tiled bf16 WMMA GEMM:
//   out[M,Ncols] = act(A[M,K](bf16, row-major) @ W(bf16, pre-swizzled) + bias)
//
// Block: 256 threads = 8 waves. Block tile 128(M) x 64(N); wave w owns rows
// [m0+16w, m0+16w+16) and 4 N-subtiles of 16. K step 32.
//
// A fragment (ISA 16-bit A 16x32 layout): lane(l) -> row ml=l&15, group g=l>>4;
//   elements 0..7  = K (k0 + 8g .. 8g+7)   -> one aligned b128 global load
//   elements 8..15 = K (k0 + 16+8g ..)     -> one aligned b128 global load
// B fragment: lane -> col ml, group g; same K order as A (8g+e, 16+8g+e)
//   from the swizzled layout [ktile][ntile][n(16)][k(32)]
//   -> two aligned b128 global loads. Weights are L2-resident; no LDS, no
//   barriers, no divergence around the WMMAs (EXEC all-ones guaranteed).
// ---------------------------------------------------------------------------
template <int ACT>
__global__ __launch_bounds__(256) void gemm_bf16_wmma(
    const __bf16* __restrict__ A, const __bf16* __restrict__ Bsw,
    const float* __restrict__ bias,
    float* __restrict__ outF, __bf16* __restrict__ outB,
    int M, int K, int Ncols, int ntiles_pad)
{
    const int tid  = threadIdx.x;
    const int lane = tid & 31;
    const int wave = tid >> 5;
    const int m0   = blockIdx.y * 128;
    const int n0   = blockIdx.x * 64;
    const int g    = lane >> 4;   // half-wave group (0/1)
    const int ml   = lane & 15;

    // A row for this lane; clamp (not mask) OOB rows so no divergence occurs.
    int arow = m0 + wave * 16 + ml;
    if (arow >= M) arow = M - 1;
    const __bf16* pA = A + (size_t)arow * K + g * 8;
    // B: swizzled tile base for this block's first N-subtile.
    const __bf16* pB = Bsw + (size_t)(n0 >> 4) * 512 + ml * 32 + g * 8;
    const size_t bstep = (size_t)ntiles_pad * 512;   // advance one ktile

    v8f zero = {};
    v8f acc[4];
#pragma unroll
    for (int i = 0; i < 4; ++i) acc[i] = zero;

    for (int kt = K >> 5; kt > 0; --kt) {
        FragAB fa;
        fa.q[0] = *(const uint4*)(pA);
        fa.q[1] = *(const uint4*)(pA + 16);
#pragma unroll
        for (int nt = 0; nt < 4; ++nt) {
            FragAB fb;
            const __bf16* pb = pB + (size_t)nt * 512;
            fb.q[0] = *(const uint4*)(pb);
            fb.q[1] = *(const uint4*)(pb + 16);
            acc[nt] = __builtin_amdgcn_wmma_f32_16x16x32_bf16(
                false, fa.v, false, fb.v, (short)0, acc[nt], false, false);
        }
        pA += 32;
        pB += bstep;
    }

    // Epilogue. C/D layout: lane<16 -> M=v, lane>=16 -> M=v+8; N = lane&15.
#pragma unroll
    for (int nt = 0; nt < 4; ++nt) {
        const int n = n0 + nt * 16 + ml;
        if (n >= Ncols) continue;
        const float b = bias ? bias[n] : 0.f;
#pragma unroll
        for (int v = 0; v < 8; ++v) {
            const int m = m0 + wave * 16 + g * 8 + v;
            if (m >= M) continue;
            float val = acc[nt][v] + b;
            if (ACT == 1) val = elu1(val);
            if (outF) outF[(size_t)m * Ncols + n] = val;
            if (outB) outB[(size_t)m * Ncols + n] = (__bf16)val;
        }
    }
}

// ---------------------------------------------------------------------------
// Weight swizzle: W f32 [K, Ncols] -> bf16 fragment-native layout
//   out[((ktile*ntiles_pad + ntile)*16 + nl)*32 + kk],
//   k = ktile*32+kk (zero-padded to Kpad), n = ntile*16+nl (zero-padded).
// ---------------------------------------------------------------------------
__global__ void swizzle_w(const float* __restrict__ W, __bf16* __restrict__ out,
                          int K, int Kpad, int Ncols, int ntiles_pad, int total)
{
    const int idx = blockIdx.x * blockDim.x + threadIdx.x;
    if (idx >= total) return;
    const int kk    = idx & 31;
    const int nl    = (idx >> 5) & 15;
    const int tile  = idx >> 9;
    const int ntile = tile % ntiles_pad;
    const int ktile = tile / ntiles_pad;
    const int k = ktile * 32 + kk;
    const int n = ntile * 16 + nl;
    float v = 0.f;
    if (k < K && n < Ncols) v = W[(size_t)k * Ncols + n];
    out[idx] = (__bf16)v;
}

// ---------------------------------------------------------------------------
// Input prep
// ---------------------------------------------------------------------------
__global__ void build_xin(const float* __restrict__ x, const float* __restrict__ v,
                          __bf16* __restrict__ out, int N)
{
    const int idx = blockIdx.x * blockDim.x + threadIdx.x;
    if (idx >= N * 64) return;
    const int n = idx >> 6, c = idx & 63;
    float val = 0.f;
    if (c < 30)       val = x[n * 30 + c];
    else if (c < 60)  val = v[n * 30 + (c - 30)];
    out[idx] = (__bf16)val;
}

__global__ void init_rowbias(float* __restrict__ out, const float* __restrict__ bias,
                             int total, int C)
{
    const int idx = blockIdx.x * blockDim.x + threadIdx.x;
    if (idx < total) out[idx] = bias[idx % C];
}

__global__ void zero_u32(unsigned* __restrict__ p, int n)
{
    const int i = blockIdx.x * blockDim.x + threadIdx.x;
    if (i < n) p[i] = 0u;
}
__global__ void zero_f32(float* __restrict__ p, int n)
{
    const int i = blockIdx.x * blockDim.x + threadIdx.x;
    if (i < n) p[i] = 0.f;
}

// ---------------------------------------------------------------------------
// GATv2 edge passes.  Edge e < E comes from edge_index; e >= E is self loop e-E.
// ---------------------------------------------------------------------------
__global__ __launch_bounds__(256) void edge_logits(
    const float* __restrict__ xl, const float* __restrict__ xr,
    const float* __restrict__ att, const int* __restrict__ ei,
    int E, int Etot, int H, int C,
    float* __restrict__ logit, unsigned* __restrict__ lmax)
{
    const int wid  = (blockIdx.x * blockDim.x + threadIdx.x) >> 5;
    const int lane = threadIdx.x & 31;
    if (wid >= Etot * H) return;
    const int e = wid / H, h = wid % H;
    const int src = (e < E) ? ei[e] : (e - E);
    const int dst = (e < E) ? ei[E + e] : (e - E);
    const size_t HC = (size_t)H * C;
    const float* pl = xl + (size_t)src * HC + (size_t)h * C;
    const float* pr = xr + (size_t)dst * HC + (size_t)h * C;
    const float* pa = att + h * C;
    float acc = 0.f;
    for (int c = lane; c < C; c += 32) {
        float m = pl[c] + pr[c];
        m = m > 0.f ? m : 0.2f * m;          // leaky_relu(0.2)
        acc += m * pa[c];
    }
    for (int off = 16; off; off >>= 1) acc += __shfl_down(acc, off, 32);
    if (lane == 0) {
        logit[wid] = acc;
        atomicMax(lmax + (size_t)dst * H + h, f2key(acc));
    }
}

__global__ void edge_exp(const int* __restrict__ ei, int E, int Etot, int H,
                         float* __restrict__ logit /* -> ex in place */,
                         const unsigned* __restrict__ lmax, float* __restrict__ den)
{
    const int i = blockIdx.x * blockDim.x + threadIdx.x;
    if (i >= Etot * H) return;
    const int e = i / H, h = i % H;
    const int dst = (e < E) ? ei[E + e] : (e - E);
    const float ex = expf(logit[i] - key2f(lmax[(size_t)dst * H + h]));
    logit[i] = ex;
    __hip_atomic_fetch_add(den + (size_t)dst * H + h, ex,
                           __ATOMIC_RELAXED, __HIP_MEMORY_SCOPE_AGENT);
}

__global__ __launch_bounds__(256) void edge_aggregate(
    const float* __restrict__ xl, const int* __restrict__ ei,
    int E, int Etot, int H, int C,
    const float* __restrict__ ex, const float* __restrict__ den,
    float* __restrict__ out)
{
    const int wid  = (blockIdx.x * blockDim.x + threadIdx.x) >> 5;
    const int lane = threadIdx.x & 31;
    if (wid >= Etot * H) return;
    const int e = wid / H, h = wid % H;
    const int src = (e < E) ? ei[e] : (e - E);
    const int dst = (e < E) ? ei[E + e] : (e - E);
    const size_t HC = (size_t)H * C;
    const float alpha = ex[wid] / den[(size_t)dst * H + h];
    const float* pl = xl + (size_t)src * HC + (size_t)h * C;
    float* po = out + (size_t)dst * HC + (size_t)h * C;
    for (int c = lane; c < C; c += 32)
        __hip_atomic_fetch_add(po + c, alpha * pl[c],
                               __ATOMIC_RELAXED, __HIP_MEMORY_SCOPE_AGENT);
}

// ---------------------------------------------------------------------------
// BatchNorm (training-mode batch stats, biased variance) + ELU + bf16 cast
// ---------------------------------------------------------------------------
__global__ __launch_bounds__(256) void bn_stats(
    const float* __restrict__ h, int N, int C,
    float* __restrict__ mu, float* __restrict__ rs)
{
    __shared__ float ss[256], sq[256];
    const int c  = blockIdx.x * 64 + (threadIdx.x & 63);
    const int r0 = threadIdx.x >> 6;   // 0..3
    float s = 0.f, q = 0.f;
    if (c < C) {
        for (int r = r0; r < N; r += 4) {
            const float v = h[(size_t)r * C + c];
            s += v; q += v * v;
        }
    }
    ss[threadIdx.x] = s; sq[threadIdx.x] = q;
    __syncthreads();
    if (threadIdx.x < 64 && c < C) {
        const float S = ss[threadIdx.x] + ss[threadIdx.x + 64] +
                        ss[threadIdx.x + 128] + ss[threadIdx.x + 192];
        const float Q = sq[threadIdx.x] + sq[threadIdx.x + 64] +
                        sq[threadIdx.x + 128] + sq[threadIdx.x + 192];
        const float m   = S / (float)N;
        const float var = Q / (float)N - m * m;
        mu[c] = m;
        rs[c] = rsqrtf(var + 1e-5f);
    }
}

__global__ void bn_apply_elu(const float* __restrict__ h,
                             const float* __restrict__ mu, const float* __restrict__ rs,
                             const float* __restrict__ g,  const float* __restrict__ be,
                             __bf16* __restrict__ out, int total, int C)
{
    const int idx = blockIdx.x * blockDim.x + threadIdx.x;
    if (idx >= total) return;
    const int c = idx % C;
    float y = (h[idx] - mu[c]) * rs[c] * g[c] + be[c];
    out[idx] = (__bf16)elu1(y);
}

// ---------------------------------------------------------------------------
// Host orchestration
// ---------------------------------------------------------------------------
static inline int cdiv(int a, int b) { return (a + b - 1) / b; }

extern "C" void kernel_launch(void* const* d_in, const int* in_sizes, int n_in,
                              void* d_out, int out_size, void* d_ws, size_t ws_size,
                              hipStream_t stream)
{
    const float* x_pca = (const float*)d_in[0];
    const float* v_pca = (const float*)d_in[1];
    const float* W1_l  = (const float*)d_in[2];
    const float* b1_l  = (const float*)d_in[3];
    const float* W1_r  = (const float*)d_in[4];
    const float* b1_r  = (const float*)d_in[5];
    const float* att1  = (const float*)d_in[6];
    const float* bias1 = (const float*)d_in[7];
    const float* g1    = (const float*)d_in[8];
    const float* be1   = (const float*)d_in[9];
    const float* W2_l  = (const float*)d_in[10];
    const float* b2_l  = (const float*)d_in[11];
    const float* W2_r  = (const float*)d_in[12];
    const float* b2_r  = (const float*)d_in[13];
    const float* att2  = (const float*)d_in[14];
    const float* bias2 = (const float*)d_in[15];
    const float* g2    = (const float*)d_in[16];
    const float* be2   = (const float*)d_in[17];
    const float* Wd1   = (const float*)d_in[18];
    const float* bd1   = (const float*)d_in[19];
    const float* Wd2   = (const float*)d_in[20];
    const float* bd2   = (const float*)d_in[21];
    const float* Wd3   = (const float*)d_in[22];
    const float* bd3   = (const float*)d_in[23];
    const int*   ei    = (const int*)d_in[24];

    const int N    = in_sizes[0] / 30;   // 30000
    const int E    = in_sizes[24] / 2;   // 240000
    const int Etot = E + N;              // with self loops

    // ---- workspace bump allocator (256B aligned) ----
    char* ws = (char*)d_ws;
    size_t off = 0;
    auto alloc = [&](size_t bytes) -> char* {
        off = (off + 255) & ~(size_t)255;
        char* p = ws + off;
        off += bytes;
        return p;
    };

    // swizzled-weight geometry: ntiles padded to multiple of 4 (64 cols/block)
    const int nt1  = cdiv(512, 64) * 4;    // 32   (W1_l/r: 64x512)
    const int nt2  = cdiv(128, 64) * 4;    // 8    (W2_l/r: 512x128)
    const int ntd1 = cdiv(512, 64) * 4;    // 32   (Wd1: 128x512)
    const int ntd2 = cdiv(1024, 64) * 4;   // 64   (Wd2: 512x1024)
    const int ntd3 = cdiv(2000, 64) * 4;   // 128  (Wd3: 1024x2000 -> 2048 pad)
    const int sz1  = (64 / 32)   * nt1  * 512;
    const int sz2  = (512 / 32)  * nt2  * 512;
    const int szd1 = (128 / 32)  * ntd1 * 512;
    const int szd2 = (512 / 32)  * ntd2 * 512;
    const int szd3 = (1024 / 32) * ntd3 * 512;

    __bf16* x_in  = (__bf16*)alloc((size_t)N * 64 * 2);
    __bf16* w1l   = (__bf16*)alloc((size_t)sz1 * 2);
    __bf16* w1r   = (__bf16*)alloc((size_t)sz1 * 2);
    __bf16* w2l   = (__bf16*)alloc((size_t)sz2 * 2);
    __bf16* w2r   = (__bf16*)alloc((size_t)sz2 * 2);
    __bf16* wd1   = (__bf16*)alloc((size_t)szd1 * 2);
    __bf16* wd2   = (__bf16*)alloc((size_t)szd2 * 2);
    __bf16* wd3   = (__bf16*)alloc((size_t)szd3 * 2);
    float*  xl1   = (float*)alloc((size_t)N * 512 * 4);   // later reused as d2 (bf16)
    float*  xr1   = (float*)alloc((size_t)N * 512 * 4);   // later reused as d1 (bf16)
    float*  out1  = (float*)alloc((size_t)N * 512 * 4);   // later reused as xl2/xr2/out2
    __bf16* h1    = (__bf16*)alloc((size_t)N * 512 * 2);
    __bf16* h2    = (__bf16*)alloc((size_t)N * 128 * 2);
    float*  logit1= (float*)alloc((size_t)Etot * 4 * 4);
    unsigned* lmax1 = (unsigned*)alloc((size_t)N * 4 * 4);
    float*  den1  = (float*)alloc((size_t)N * 4 * 4);
    float*  logit2= (float*)alloc((size_t)Etot * 4);
    unsigned* lmax2 = (unsigned*)alloc((size_t)N * 4);
    float*  den2  = (float*)alloc((size_t)N * 4);
    float*  mu1   = (float*)alloc(512 * 4);
    float*  rs1   = (float*)alloc(512 * 4);
    float*  mu2   = (float*)alloc(128 * 4);
    float*  rs2   = (float*)alloc(128 * 4);
    (void)ws_size; (void)n_in;

    // aliases (dead-buffer reuse)
    float*  xl2 = out1;                    // [N,128]
    float*  xr2 = out1 + (size_t)N * 128;  // [N,128]
    float*  out2= out1 + (size_t)N * 256;  // [N,128]
    __bf16* d1  = (__bf16*)xr1;            // [N,512]
    __bf16* d2  = (__bf16*)xl1;            // [N,1024]

    auto gemm = [&](int act, const __bf16* A, const __bf16* B, const float* bias,
                    float* oF, __bf16* oB, int M, int K, int Ncols, int ntp) {
        dim3 grid(cdiv(Ncols, 64), cdiv(M, 128));
        if (act)
            hipLaunchKernelGGL((gemm_bf16_wmma<1>), grid, dim3(256), 0, stream,
                               A, B, bias, oF, oB, M, K, Ncols, ntp);
        else
            hipLaunchKernelGGL((gemm_bf16_wmma<0>), grid, dim3(256), 0, stream,
                               A, B, bias, oF, oB, M, K, Ncols, ntp);
    };
    auto swz = [&](const float* W, __bf16* o, int K, int Kpad, int Nc, int ntp, int total) {
        swizzle_w<<<cdiv(total, 256), 256, 0, stream>>>(W, o, K, Kpad, Nc, ntp, total);
    };

    // ---- prep: x_in bf16 [N,64]; weights -> swizzled bf16 ----
    build_xin<<<cdiv(N * 64, 256), 256, 0, stream>>>(x_pca, v_pca, x_in, N);
    swz(W1_l, w1l, 60, 64, 512, nt1, sz1);
    swz(W1_r, w1r, 60, 64, 512, nt1, sz1);
    swz(W2_l, w2l, 512, 512, 128, nt2, sz2);
    swz(W2_r, w2r, 512, 512, 128, nt2, sz2);
    swz(Wd1, wd1, 128, 128, 512, ntd1, szd1);
    swz(Wd2, wd2, 512, 512, 1024, ntd2, szd2);
    swz(Wd3, wd3, 1024, 1024, 2000, ntd3, szd3);

    // ---- layer 1: xl = x_in@W1_l + b1_l ; xr = x_in@W1_r + b1_r ----
    gemm(0, x_in, w1l, b1_l, xl1, nullptr, N, 64, 512, nt1);
    gemm(0, x_in, w1r, b1_r, xr1, nullptr, N, 64, 512, nt1);

    zero_u32<<<cdiv(N * 4, 256), 256, 0, stream>>>(lmax1, N * 4);
    zero_f32<<<cdiv(N * 4, 256), 256, 0, stream>>>(den1, N * 4);
    init_rowbias<<<cdiv(N * 512, 256), 256, 0, stream>>>(out1, bias1, N * 512, 512);

    {
        const int tasks = Etot * 4;                 // (edge, head) pairs
        const int wblocks = cdiv(tasks * 32, 256);  // one wave per task
        edge_logits<<<wblocks, 256, 0, stream>>>(xl1, xr1, att1, ei, E, Etot, 4, 128,
                                                 logit1, lmax1);
        edge_exp<<<cdiv(tasks, 256), 256, 0, stream>>>(ei, E, Etot, 4, logit1, lmax1, den1);
        edge_aggregate<<<wblocks, 256, 0, stream>>>(xl1, ei, E, Etot, 4, 128,
                                                    logit1, den1, out1);
    }
    bn_stats<<<cdiv(512, 64), 256, 0, stream>>>(out1, N, 512, mu1, rs1);
    bn_apply_elu<<<cdiv(N * 512, 256), 256, 0, stream>>>(out1, mu1, rs1, g1, be1,
                                                         h1, N * 512, 512);

    // ---- layer 2 ----
    gemm(0, h1, w2l, b2_l, xl2, nullptr, N, 512, 128, nt2);
    gemm(0, h1, w2r, b2_r, xr2, nullptr, N, 512, 128, nt2);

    zero_u32<<<cdiv(N, 256), 256, 0, stream>>>(lmax2, N);
    zero_f32<<<cdiv(N, 256), 256, 0, stream>>>(den2, N);
    init_rowbias<<<cdiv(N * 128, 256), 256, 0, stream>>>(out2, bias2, N * 128, 128);

    {
        const int tasks = Etot;                     // heads = 1
        const int wblocks = cdiv(tasks * 32, 256);
        edge_logits<<<wblocks, 256, 0, stream>>>(xl2, xr2, att2, ei, E, Etot, 1, 128,
                                                 logit2, lmax2);
        edge_exp<<<cdiv(tasks, 256), 256, 0, stream>>>(ei, E, Etot, 1, logit2, lmax2, den2);
        edge_aggregate<<<wblocks, 256, 0, stream>>>(xl2, ei, E, Etot, 1, 128,
                                                    logit2, den2, out2);
    }
    bn_stats<<<cdiv(128, 64), 256, 0, stream>>>(out2, N, 128, mu2, rs2);
    bn_apply_elu<<<cdiv(N * 128, 256), 256, 0, stream>>>(out2, mu2, rs2, g2, be2,
                                                         h2, N * 128, 128);

    // ---- decoder ----
    gemm(1, h2, wd1, bd1, nullptr, d1, N, 128, 512, ntd1);
    gemm(1, d1, wd2, bd2, nullptr, d2, N, 512, 1024, ntd2);
    gemm(0, d2, wd3, bd3, (float*)d_out, nullptr, N, 1024, 2000, ntd3);
    (void)out_size;
}
